// SelfAttention_90804198572405
// MI455X (gfx1250) — matrix operand order repaired and checked by reading, not compiled, not run on hardware
//
#include <hip/hip_runtime.h>
#include <hip/hip_bf16.h>
#include <stdint.h>

#define NUM_HEADS 16
#define HEAD_DIM  128
#define HDIM      2048
#define ROTARY_DIM 32
#define ATT_SCALE 0.08838834764831845f  /* 128^-0.5 */

typedef __bf16 bf16_t;
typedef __attribute__((ext_vector_type(16))) __bf16 v16bf;
typedef __attribute__((ext_vector_type(8)))  __bf16 v8bf;
typedef __attribute__((ext_vector_type(8)))  float  v8f;
typedef __attribute__((ext_vector_type(4)))  unsigned u32x4;
typedef __attribute__((ext_vector_type(4)))  int      i32x4;
typedef __attribute__((ext_vector_type(8)))  int      i32x8;

static __device__ __forceinline__ unsigned short f2bfu(float f) {
  union { float f; unsigned u; } v; v.f = f;
  unsigned r = v.u + 0x7FFFu + ((v.u >> 16) & 1u);
  return (unsigned short)(r >> 16);
}
static __device__ __forceinline__ bf16_t f2bf(float f) {
  unsigned short h = f2bfu(f);
  return __builtin_bit_cast(bf16_t, h);
}
static __device__ __forceinline__ v16bf cat16(v8bf lo, v8bf hi) {
  return __builtin_shufflevector(lo, hi, 0,1,2,3,4,5,6,7,8,9,10,11,12,13,14,15);
}
static __device__ __forceinline__ v8f wmma_bf16(v16bf a, v16bf b, v8f c) {
  // D = A(16x32) * B(32x16) + C, f32 accumulate
  return __builtin_amdgcn_wmma_f32_16x16x32_bf16(false, a, false, b, (short)0, c,
                                                 false, false);
}

// --- CDNA5 async copy: global -> LDS, 16B per lane, tracked by ASYNCcnt ----
static __device__ __forceinline__ unsigned lds_addr(const void* p) {
  // low 32 bits of a flat shared-aperture address == LDS byte offset
  return (unsigned)(uintptr_t)p;
}
static __device__ __forceinline__ void async_load_b128(const void* gptr,
                                                       unsigned lds_off) {
  asm volatile("global_load_async_to_lds_b128 %0, %1, off"
               :: "v"(lds_off), "v"(gptr)
               : "memory");
}
#if __has_builtin(__builtin_amdgcn_s_wait_asynccnt)
#define WAIT_ASYNCCNT(n) __builtin_amdgcn_s_wait_asynccnt(n)
#else
#define WAIT_ASYNCCNT(n) asm volatile("s_wait_asynccnt %0" :: "n"(n) : "memory")
#endif

// --- CDNA5 Tensor Data Mover: one-descriptor 2-D tile copy global -> LDS ---
#if __has_builtin(__builtin_amdgcn_tensor_load_to_lds)
#define HAVE_TDM 1
// D# per CDNA5 ISA (2-D tensor, groups 2/3
// zero).  data_size = 2 bytes (code 1); tile walks X (tile_d0) then Y
// (tile_d1), writing LDS contiguously -> matches [tile_d1][tile_d0] row-major.
static __device__ __forceinline__ void tdm_load_2d(
    const void* gtile, unsigned lds_off, unsigned tensor_d0, unsigned tensor_d1,
    unsigned tile_d0, unsigned tile_d1, unsigned row_stride_elems) {
  const unsigned long long ga = (unsigned long long)(uintptr_t)gtile;
  u32x4 g0;
  g0[0] = 1u;                                   // count=1 (valid), user mode
  g0[1] = lds_off;                              // lds_addr (bytes)
  g0[2] = (unsigned)ga;                         // global_addr[31:0]
  g0[3] = (unsigned)(ga >> 32) | (2u << 30);    // global_addr[56:32] | type=2
  i32x8 g1;
  g1[0] = (int)(1u << 16);                      // wg_mask=0, data_size=2B
  g1[1] = (int)((tensor_d0 & 0xFFFFu) << 16);   // tensor_dim0[15:0] @ bits63:48
  g1[2] = (int)((tensor_d0 >> 16) | ((tensor_d1 & 0xFFFFu) << 16));
  g1[3] = (int)((tensor_d1 >> 16) | (tile_d0 << 16));   // tile_dim0 @ 127:112
  g1[4] = (int)(tile_d1 & 0xFFFFu);             // tile_dim1; tile_dim2=0
  g1[5] = (int)row_stride_elems;                // tensor_dim0_stride[31:0]
  g1[6] = 0;                                    // stride hi / dim1_stride lo
  g1[7] = 0;
  const i32x4 z4 = {0, 0, 0, 0};
#if __clang_major__ >= 23
  // amdgpu-toolchain / clang-23: 6-arg form (extra int32x8 before cpol)
  const i32x8 z8 = {0, 0, 0, 0, 0, 0, 0, 0};
  __builtin_amdgcn_tensor_load_to_lds(g0, g1, z4, z4, z8, 0);
#else
  // ROCm 7.2 / clang-22: 5-arg form
  __builtin_amdgcn_tensor_load_to_lds(g0, g1, z4, z4, 0);
#endif
}
#define WAIT_TENSORCNT(n) __builtin_amdgcn_s_wait_tensorcnt(n)
#else
#define HAVE_TDM 0
#endif

// ---------------------------------------------------------------------------
// Kernel 1a: fp32 -> bf16 flat conversion (vectorized x4)
// ---------------------------------------------------------------------------
__global__ void cvt_f32_bf16(const float* __restrict__ src,
                             bf16_t* __restrict__ dst, int n4) {
  int i = blockIdx.x * blockDim.x + threadIdx.x;
  if (i >= n4) return;
  const float4 v = ((const float4*)src)[i];
  ushort4 o;
  o.x = f2bfu(v.x); o.y = f2bfu(v.y); o.z = f2bfu(v.z); o.w = f2bfu(v.w);
  ((ushort4*)dst)[i] = o;
}

// ---------------------------------------------------------------------------
// Kernel 1b: fp32 [K][N] -> bf16 [N][K] transpose (32x32 LDS tiles)
// ---------------------------------------------------------------------------
__global__ __launch_bounds__(256) void cvt_transpose_bf16(
    const float* __restrict__ src, bf16_t* __restrict__ dst, int K, int N) {
  __shared__ float tile[32][33];
  const int k0 = blockIdx.y * 32, n0 = blockIdx.x * 32;
  const int tx = threadIdx.x & 31, ty = threadIdx.x >> 5;  // ty = 0..7
  #pragma unroll
  for (int r = 0; r < 32; r += 8)
    tile[ty + r][tx] = src[(size_t)(k0 + ty + r) * N + n0 + tx];
  __syncthreads();
  #pragma unroll
  for (int r = 0; r < 32; r += 8)
    dst[(size_t)(n0 + ty + r) * K + k0 + tx] = f2bf(tile[tx][ty + r]);
}

// ---------------------------------------------------------------------------
// Kernel 2/5: C[M,N](f32) = A[M,K](bf16) * Bt[N,K](bf16, pre-transposed)
// Block 256 = 8 waves; block tile 64x256, wave tile 32x64.
// A tile double-buffered in LDS via the Tensor Data Mover (one descriptor per
// K-step, issued by wave 0, TENSORcnt-tracked); falls back to per-lane
// global_load_async_to_lds_b128 if the TDM builtin is unavailable.
// B fragments contiguous 32B per lane straight from global (L2-resident).
// Requires M%64==0, N%256==0, K%32==0.
// ---------------------------------------------------------------------------
__global__ __launch_bounds__(256) void gemm_bf16_tn(
    const bf16_t* __restrict__ A, const bf16_t* __restrict__ Bt,
    float* __restrict__ C, int M, int N, int K) {
  __shared__ alignas(16) bf16_t As[2][64][32];   // 2 x 4 KB
  const int tid = threadIdx.x;
  const int wid = tid >> 5, lane = tid & 31;
  const int hlf = lane >> 4, l16 = lane & 15;
  const int m0 = blockIdx.y * 64;
  const int n0 = blockIdx.x * 256;
  const int wm = (wid >> 2) * 32;   // wave row offset (0/32)
  const int wn = (wid & 3) * 64;    // wave col offset (0..192)
  v8f acc[2][4] = {};
  const bf16_t* atile = A + (size_t)m0 * K;
#if !HAVE_TDM
  const int ar = tid >> 2, ac = (tid & 3) * 8;   // A staging map: 1x16B/thread
  const bf16_t* agp = atile + (size_t)ar * K + ac;
#endif
  const bf16_t* bp[4];
  #pragma unroll
  for (int ni = 0; ni < 4; ++ni)
    bp[ni] = Bt + (size_t)(n0 + wn + ni * 16 + l16) * K + hlf * 16;

  const int nkt = K >> 5;
#if HAVE_TDM
  if (wid == 0)
    tdm_load_2d(atile, lds_addr(&As[0][0][0]), K, 64, 32, 64, K);
#else
  async_load_b128(agp, lds_addr(&As[0][ar][ac]));
#endif
  for (int kt = 0; kt < nkt; ++kt) {
    const int cur = kt & 1;
    const int kk = kt << 5;
    __syncthreads();  // everyone done reading As[cur^1]
#if HAVE_TDM
    if (wid == 0) {
      if (kt + 1 < nkt) {
        tdm_load_2d(atile + kk + 32, lds_addr(&As[cur ^ 1][0][0]),
                    K, 64, 32, 64, K);
        WAIT_TENSORCNT(1);   // tile kt landed
      } else {
        WAIT_TENSORCNT(0);
      }
    }
#else
    if (kt + 1 < nkt) {
      async_load_b128(agp + kk + 32, lds_addr(&As[cur ^ 1][ar][ac]));
      WAIT_ASYNCCNT(1);
    } else {
      WAIT_ASYNCCNT(0);
    }
#endif
    __syncthreads();     // tile kt visible to all waves
    v16bf af[2];
    #pragma unroll
    for (int mi = 0; mi < 2; ++mi) {
      const bf16_t* p = &As[cur][wm + mi * 16 + l16][hlf * 16];
      af[mi] = cat16(*(const v8bf*)p, *(const v8bf*)(p + 8));
    }
    #pragma unroll
    for (int ni = 0; ni < 4; ++ni) {
      __builtin_prefetch(bp[ni] + kk + 32, 0, 3);  // next K-tile of B
      v16bf bfr = cat16(*(const v8bf*)(bp[ni] + kk),
                        *(const v8bf*)(bp[ni] + kk + 8));
      #pragma unroll
      for (int mi = 0; mi < 2; ++mi)
        acc[mi][ni] = wmma_bf16(af[mi], bfr, acc[mi][ni]);
    }
  }
  // epilogue: C layout M = i + 8*half, N = lane%16; one addr per 4 stores
  #pragma unroll
  for (int mi = 0; mi < 2; ++mi)
    #pragma unroll
    for (int i = 0; i < 8; ++i) {
      const int gr = m0 + wm + mi * 16 + i + 8 * hlf;
      float* rowp = C + (size_t)gr * N + n0 + wn + l16;
      #pragma unroll
      for (int ni = 0; ni < 4; ++ni) rowp[ni * 16] = acc[mi][ni][i];
    }
}

// ---------------------------------------------------------------------------
// Kernel 3: split qkv, apply RoPE to q/k, emit bf16 head-major layouts.
// Qh,Kh: [B*NH][S][128];  Vt: [B*NH][128][S] (transposed for PV WMMA B-frags)
// ---------------------------------------------------------------------------
__global__ void rope_split(const float* __restrict__ qkv,
                           bf16_t* __restrict__ Qh, bf16_t* __restrict__ Kh,
                           bf16_t* __restrict__ Vt, int S) {
  const int tid = blockIdx.x * blockDim.x + threadIdx.x;
  const int d  = tid & (HEAD_DIM - 1);
  const int s  = (tid >> 7) % S;
  const int bh = (tid >> 7) / S;
  const int b  = bh >> 4;       // /NUM_HEADS
  const int hh = bh & 15;
  const size_t row  = (size_t)b * S + s;
  const size_t base = row * (3 * HDIM) + (size_t)hh * HEAD_DIM;
  float q = qkv[base + d];
  float k = qkv[base + HDIM + d];
  float v = qkv[base + 2 * HDIM + d];
  if (d < ROTARY_DIM) {
    const int di = d & 15;
    const float inv = __expf(-9.2103403719761836f * (float)di * (1.0f / 16.0f));
    const float ang = (float)s * inv;
    const float c = __cosf(ang), sn = __sinf(ang);
    if (d < 16) {
      q = q * c - qkv[base + d + 16] * sn;
      k = k * c - qkv[base + HDIM + d + 16] * sn;
    } else {
      q = q * c + qkv[base + d - 16] * sn;
      k = k * c + qkv[base + HDIM + d - 16] * sn;
    }
  }
  const size_t hrow = (size_t)bh * S + s;
  Qh[hrow * HEAD_DIM + d] = f2bf(q);
  Kh[hrow * HEAD_DIM + d] = f2bf(k);
  Vt[((size_t)bh * HEAD_DIM + d) * S + s] = f2bf(v);
}

// ---------------------------------------------------------------------------
// Kernel 4: flash attention.  grid = (S/128, B*NH), 8 waves/block, each wave
// owns 16 q-rows.  K/V tiles (32 keys) shared by all waves, double-buffered
// in LDS via async copies (ASYNCcnt).  Online softmax, fp32 accumulators.
// ---------------------------------------------------------------------------
__global__ __launch_bounds__(256) void attn_flash(
    const bf16_t* __restrict__ Qh, const bf16_t* __restrict__ Kh,
    const bf16_t* __restrict__ Vt, const float* __restrict__ mask,
    bf16_t* __restrict__ Ctx, int S) {
  __shared__ alignas(16) bf16_t Ks[2][32][128];  // 16 KB, row-major per key
  __shared__ alignas(16) bf16_t Vs[2][128][32];  // 16 KB, d-major
  __shared__ alignas(16) bf16_t Ps[8][16][32];   //  8 KB, per-wave P staging
  const int tid = threadIdx.x;
  const int wid = tid >> 5, lane = tid & 31;
  const int hlf = lane >> 4, l16 = lane & 15;
  const int bh = blockIdx.y, b = bh >> 4, hh = bh & 15;
  const int q0 = blockIdx.x * 128 + wid * 16;
  const bf16_t* Qp = Qh + (size_t)bh * S * HEAD_DIM;
  const bf16_t* Kp = Kh + (size_t)bh * S * HEAD_DIM;
  const bf16_t* Vp = Vt + (size_t)bh * HEAD_DIM * S;
  const float*  Mp = mask + (size_t)b * S * S;

  // staging maps (4 x 16B async per thread per 32-key tile)
  const int kr = tid >> 3, kc = (tid & 7) * 16;   // K: 32 rows x 128 d
  const int vd = tid >> 1, vc = (tid & 1) * 16;   // V: 128 d x 32 keys
  const bf16_t* kgp = Kp + (size_t)kr * HEAD_DIM + kc;
  const bf16_t* vgp = Vp + (size_t)vd * S + vc;

  // Q resident as 4 A-fragments covering D=128
  v16bf qf[4];
  {
    const int qrow = q0 + l16;
    #pragma unroll
    for (int c = 0; c < 4; ++c) {
      const bf16_t* p = Qp + (size_t)qrow * HEAD_DIM + 32 * c + hlf * 16;
      qf[c] = cat16(*(const v8bf*)p, *(const v8bf*)(p + 8));
    }
  }
  v8f o[8] = {};
  float m_i[8], l_i[8];
  #pragma unroll
  for (int i = 0; i < 8; ++i) { m_i[i] = -3.0e38f; l_i[i] = 0.0f; }

  // prologue: stage tile 0
  async_load_b128(kgp,     lds_addr(&Ks[0][kr][kc]));
  async_load_b128(kgp + 8, lds_addr(&Ks[0][kr][kc + 8]));
  async_load_b128(vgp,     lds_addr(&Vs[0][vd][vc]));
  async_load_b128(vgp + 8, lds_addr(&Vs[0][vd][vc + 8]));

  const int nkt = S >> 5;
  for (int kt = 0; kt < nkt; ++kt) {
    const int k0 = kt << 5;
    const int cur = kt & 1;
    if (kt + 1 < nkt) {
      __syncthreads();  // everyone done reading buffer cur^1
      const bf16_t* kn = kgp + (size_t)(k0 + 32) * HEAD_DIM;
      const bf16_t* vn = vgp + (k0 + 32);
      async_load_b128(kn,     lds_addr(&Ks[cur ^ 1][kr][kc]));
      async_load_b128(kn + 8, lds_addr(&Ks[cur ^ 1][kr][kc + 8]));
      async_load_b128(vn,     lds_addr(&Vs[cur ^ 1][vd][vc]));
      async_load_b128(vn + 8, lds_addr(&Vs[cur ^ 1][vd][vc + 8]));
      WAIT_ASYNCCNT(4);  // tile kt landed (this wave)
    } else {
      WAIT_ASYNCCNT(0);
    }
    __syncthreads();     // tile kt landed (all waves)

    // ---- scores: two 16-col subtiles, K-dim (=D) in 4 chunks of 32
    v8f sc[2];
    #pragma unroll
    for (int st = 0; st < 2; ++st) {
      v8f acc = {};
      const bf16_t* kb = &Ks[cur][st * 16 + l16][hlf * 16];
      #pragma unroll
      for (int c = 0; c < 4; ++c) {
        v16bf kf = cat16(*(const v8bf*)(kb + 32 * c),
                         *(const v8bf*)(kb + 32 * c + 8));
        acc = wmma_bf16(qf[c], kf, acc);
      }
      const int kcol = k0 + st * 16 + l16;
      #pragma unroll
      for (int i = 0; i < 8; ++i) {
        const int qrow = q0 + i + 8 * hlf;
        sc[st][i] = acc[i] * ATT_SCALE + Mp[(size_t)qrow * S + kcol];
      }
    }
    // ---- online softmax (rows stripe cols across 16 lanes)
    float mnew[8], alpha[8];
    #pragma unroll
    for (int i = 0; i < 8; ++i) {
      float r = fmaxf(sc[0][i], sc[1][i]);
      r = fmaxf(r, __shfl_xor(r, 1));
      r = fmaxf(r, __shfl_xor(r, 2));
      r = fmaxf(r, __shfl_xor(r, 4));
      r = fmaxf(r, __shfl_xor(r, 8));
      mnew[i]  = fmaxf(m_i[i], r);
      alpha[i] = __expf(m_i[i] - mnew[i]);
      m_i[i]   = mnew[i];
    }
    #pragma unroll
    for (int st = 0; st < 2; ++st)
      #pragma unroll
      for (int i = 0; i < 8; ++i) sc[st][i] = __expf(sc[st][i] - mnew[i]);
    #pragma unroll
    for (int i = 0; i < 8; ++i) {
      float r = sc[0][i] + sc[1][i];
      r += __shfl_xor(r, 1);
      r += __shfl_xor(r, 2);
      r += __shfl_xor(r, 4);
      r += __shfl_xor(r, 8);
      l_i[i] = l_i[i] * alpha[i] + r;
    }
    #pragma unroll
    for (int dc = 0; dc < 8; ++dc)
      #pragma unroll
      for (int i = 0; i < 8; ++i) o[dc][i] *= alpha[i];
    // ---- P: C-layout f32 -> A-layout bf16 via per-wave LDS (DS in-order)
    #pragma unroll
    for (int st = 0; st < 2; ++st)
      #pragma unroll
      for (int i = 0; i < 8; ++i)
        Ps[wid][i + 8 * hlf][st * 16 + l16] = f2bf(sc[st][i]);
    v16bf pf = cat16(*(const v8bf*)&Ps[wid][l16][hlf * 16],
                     *(const v8bf*)&Ps[wid][l16][hlf * 16 + 8]);
    // ---- O += P * V (V staged d-major: contiguous 32B per lane)
    #pragma unroll
    for (int dc = 0; dc < 8; ++dc) {
      const bf16_t* vb = &Vs[cur][dc * 16 + l16][hlf * 16];
      v16bf vf = cat16(*(const v8bf*)vb, *(const v8bf*)(vb + 8));
      o[dc] = wmma_bf16(pf, vf, o[dc]);
    }
  }
  #pragma unroll
  for (int i = 0; i < 8; ++i) l_i[i] = 1.0f / l_i[i];
  // ctx back to [B][S][H] bf16 for the output projection GEMM
  #pragma unroll
  for (int dc = 0; dc < 8; ++dc)
    #pragma unroll
    for (int i = 0; i < 8; ++i) {
      const int qrow = q0 + i + 8 * hlf;
      Ctx[((size_t)b * S + qrow) * HDIM + (size_t)hh * HEAD_DIM + dc * 16 + l16] =
          f2bf(o[dc][i] * l_i[i]);
    }
}

// ---------------------------------------------------------------------------
// Host launcher
// ---------------------------------------------------------------------------
extern "C" void kernel_launch(void* const* d_in, const int* in_sizes, int n_in,
                              void* d_out, int out_size, void* d_ws, size_t ws_size,
                              hipStream_t stream) {
  (void)n_in; (void)out_size; (void)ws_size;
  const float* hs   = (const float*)d_in[0];
  const float* msk  = (const float*)d_in[1];
  const float* Wqkv = (const float*)d_in[2];
  const float* Wout = (const float*)d_in[3];
  float* out = (float*)d_out;

  const int BS = in_sizes[0] / HDIM;                                   // B*S
  const int S  = (int)(((long long)in_sizes[1] * HDIM) / in_sizes[0]); // seq
  const int B  = BS / S;

  char* w = (char*)d_ws;
  size_t off = 0;
  auto carve = [&](size_t bytes) -> char* {
    char* p = w + off;
    off += (bytes + 255) & ~(size_t)255;
    return p;
  };
  bf16_t* hsb    = (bf16_t*)carve((size_t)BS * HDIM * 2);
  bf16_t* wqkvT  = (bf16_t*)carve((size_t)HDIM * 3 * HDIM * 2);  // [6144][2048]
  bf16_t* woutT  = (bf16_t*)carve((size_t)HDIM * HDIM * 2);      // [2048][2048]
  float*  qkv    = (float*) carve((size_t)BS * 3 * HDIM * 4);
  bf16_t* Qh     = (bf16_t*)carve((size_t)BS * HDIM * 2);
  bf16_t* Kh     = (bf16_t*)carve((size_t)BS * HDIM * 2);
  bf16_t* Vt     = (bf16_t*)carve((size_t)BS * HDIM * 2);
  bf16_t* Ctx    = (bf16_t*)carve((size_t)BS * HDIM * 2);

  {
    const int n4 = BS * HDIM / 4;
    cvt_f32_bf16<<<(n4 + 255) / 256, 256, 0, stream>>>(hs, hsb, n4);
  }
  cvt_transpose_bf16<<<dim3(3 * HDIM / 32, HDIM / 32), 256, 0, stream>>>(
      Wqkv, wqkvT, HDIM, 3 * HDIM);
  cvt_transpose_bf16<<<dim3(HDIM / 32, HDIM / 32), 256, 0, stream>>>(
      Wout, woutT, HDIM, HDIM);
  // QKV projection: [BS,2048] x [2048,6144] -> fp32
  gemm_bf16_tn<<<dim3(3 * HDIM / 256, BS / 64), 256, 0, stream>>>(
      hsb, wqkvT, qkv, BS, 3 * HDIM, HDIM);
  // RoPE + head split
  rope_split<<<(BS * HDIM) / 256, 256, 0, stream>>>(qkv, Qh, Kh, Vt, S);
  // attention
  attn_flash<<<dim3(S / 128, B * NUM_HEADS), 256, 0, stream>>>(
      Qh, Kh, Vt, msk, Ctx, S);
  // output projection: [BS,2048] x [2048,2048] -> d_out fp32
  gemm_bf16_tn<<<dim3(HDIM / 256, BS / 64), 256, 0, stream>>>(
      Ctx, woutT, out, BS, HDIM, HDIM);
}
